// TabularAttributeEncoder_67559835566111
// MI455X (gfx1250) — compile-verified, matrix operand order checked by reading, not run on hardware
//
#include <hip/hip_runtime.h>

typedef _Float16 v16h __attribute__((ext_vector_type(16)));
typedef _Float16 v8h  __attribute__((ext_vector_type(8)));
typedef float    v8f  __attribute__((ext_vector_type(8)));

#define B_SZ   8192
#define P_NUM  24
#define Q_CAT  12
#define E_DIM  256
#define V_VOC  1000
#define T_TOK  36
#define LN_EPS 1e-5f

#define MT  64        // batch rows per block
#define LDH 40        // half-element row stride for As/Bs (80B)
#define LDC 256       // float row stride for Cs epilogue tile
#define ABUF_BYTES (MT * LDH * 2)            // 5120
#define BUF_BYTES  (ABUF_BYTES + E_DIM * LDH * 2)  // 5120 + 20480 = 25600

// Branchless gelu (exact form, Abramowitz-Stegun 7.1.26 erf, |err| < 1.5e-7:
// far below the f16 quantization applied right after).
__device__ __forceinline__ float gelu_fast(float h) {
    const float x  = h * 0.70710678118654752f;
    const float ax = __builtin_fabsf(x);
    const float t  = __builtin_amdgcn_rcpf(__builtin_fmaf(0.3275911f, ax, 1.0f));
    float p = __builtin_fmaf(1.061405429f, t, -1.453152027f);
    p = __builtin_fmaf(p, t, 1.421413741f);
    p = __builtin_fmaf(p, t, -0.284496736f);
    p = __builtin_fmaf(p, t, 0.254829592f);
    p = p * t;
    const float e    = __builtin_amdgcn_exp2f(-ax * ax * 1.44269504088896f);
    const float erfa = __builtin_fmaf(-p, e, 1.0f);       // erf(|x|)
    const float er   = __builtin_copysignf(erfa, x);
    return 0.5f * h * (1.0f + er);
}

union HV { v16h v; v8h h[2]; };

// One block: p fixed, 64 batch rows, full N=256.  8 waves = 4(M) x 2(N).
__global__ __launch_bounds__(256)
void num_tok_kernel(const float* __restrict__ x_num,
                    const float* __restrict__ W1,
                    const float* __restrict__ b1,
                    const float* __restrict__ W2,
                    const float* __restrict__ b2,
                    const float* __restrict__ gamma,
                    const float* __restrict__ beta,
                    float* __restrict__ out) {
    __shared__ __align__(16) char smem[MT * LDC * 4];   // 65536 B (>= 2*BUF_BYTES)
    float* Cs = (float*)smem;                           // 64 x 256, reused after GEMM

    const int p  = blockIdx.x;            // 0..23
    const int b0 = blockIdx.y * MT;       // batch tile base

    const int tid     = threadIdx.x;
    const int lane    = tid & 31;
    const int wid     = tid >> 5;
    const int waveM   = wid & 3;              // 0..3 -> 16-row tile
    const int waveN   = wid >> 2;             // 0..1 -> 128-col half
    const int l16     = lane & 15;
    const int halfOff = (lane >> 4) << 3;     // 0 or 8 (K-half per WMMA layout)

    // staging thread mapping
    const int kqa = tid & 3;                  // k-octet 0..3
    const int ma  = tid >> 2;                 // A row 0..63
    const int ng  = tid >> 2;                 // B n-group 0..63 (rows 4*ng..)

    const float* W1p = W1 + p * E_DIM;
    const float* b1p = b1 + p * E_DIM;
    const float* W2p = W2 + (size_t)p * E_DIM * E_DIM;

    // K-invariant per-row x value (reused by every K-step)
    const float xv = x_num[(size_t)(b0 + ma) * P_NUM + p];

    v8f acc[8];
    const v8f vzero = {};
    #pragma unroll
    for (int t = 0; t < 8; ++t) acc[t] = vzero;

    // ---- staging helpers (lambdas keep everything in-registers) ----
    auto fillA = [&](int buf, int kb) {
        _Float16* As = (_Float16*)(smem + buf * BUF_BYTES);
        const float4 w0 = *(const float4*)(W1p + kb + kqa * 8);
        const float4 w1 = *(const float4*)(W1p + kb + kqa * 8 + 4);
        const float4 c0 = *(const float4*)(b1p + kb + kqa * 8);
        const float4 c1 = *(const float4*)(b1p + kb + kqa * 8 + 4);
        v8h a;
        a[0] = (_Float16)gelu_fast(__builtin_fmaf(xv, w0.x, c0.x));
        a[1] = (_Float16)gelu_fast(__builtin_fmaf(xv, w0.y, c0.y));
        a[2] = (_Float16)gelu_fast(__builtin_fmaf(xv, w0.z, c0.z));
        a[3] = (_Float16)gelu_fast(__builtin_fmaf(xv, w0.w, c0.w));
        a[4] = (_Float16)gelu_fast(__builtin_fmaf(xv, w1.x, c1.x));
        a[5] = (_Float16)gelu_fast(__builtin_fmaf(xv, w1.y, c1.y));
        a[6] = (_Float16)gelu_fast(__builtin_fmaf(xv, w1.z, c1.z));
        a[7] = (_Float16)gelu_fast(__builtin_fmaf(xv, w1.w, c1.w));
        *(v8h*)(As + ma * LDH + kqa * 8) = a;
    };
    auto fillB = [&](int buf, int kb) {
        _Float16* Bs = (_Float16*)(smem + buf * BUF_BYTES + ABUF_BYTES);
        const float* src = W2p + (size_t)(kb + kqa * 8) * E_DIM + ng * 4;
        v8h h0, h1, h2, h3;
        #pragma unroll
        for (int j = 0; j < 8; ++j) {
            const float4 w4 = *(const float4*)(src + (size_t)j * E_DIM);
            h0[j] = (_Float16)w4.x;
            h1[j] = (_Float16)w4.y;
            h2[j] = (_Float16)w4.z;
            h3[j] = (_Float16)w4.w;
        }
        _Float16* dst = Bs + (ng * 4) * LDH + kqa * 8;
        *(v8h*)(dst)           = h0;
        *(v8h*)(dst + LDH)     = h1;
        *(v8h*)(dst + 2 * LDH) = h2;
        *(v8h*)(dst + 3 * LDH) = h3;
    };

    // ---- double-buffered K loop: stage(k+1) overlaps wmma(k) ----
    fillA(0, 0);
    fillB(0, 0);
    __syncthreads();

    for (int step = 0; step < 8; ++step) {
        const int cur = step & 1;
        if (step < 7) {                 // uniform branch
            fillA(cur ^ 1, (step + 1) * 32);
            fillB(cur ^ 1, (step + 1) * 32);
        }
        const _Float16* As = (const _Float16*)(smem + cur * BUF_BYTES);
        const _Float16* Bs = (const _Float16*)(smem + cur * BUF_BYTES + ABUF_BYTES);

        HV afrag;
        {
            const _Float16* ap = As + (waveM * 16 + l16) * LDH + halfOff;
            afrag.h[0] = *(const v8h*)ap;
            afrag.h[1] = *(const v8h*)(ap + 16);
        }
        HV bfrag[2];
        {
            const _Float16* bp = Bs + (waveN * 128 + l16) * LDH + halfOff;
            bfrag[0].h[0] = *(const v8h*)bp;
            bfrag[0].h[1] = *(const v8h*)(bp + 16);
        }
        #pragma unroll
        for (int t = 0; t < 8; ++t) {
            if (t < 7) {   // prefetch next B fragment before this wmma
                const _Float16* bp = Bs + (waveN * 128 + (t + 1) * 16 + l16) * LDH + halfOff;
                bfrag[(t + 1) & 1].h[0] = *(const v8h*)bp;
                bfrag[(t + 1) & 1].h[1] = *(const v8h*)(bp + 16);
            }
            acc[t] = __builtin_amdgcn_wmma_f32_16x16x32_f16(
                false, afrag.v, false, bfrag[t & 1].v, (short)0, acc[t], false, false);
        }
        __syncthreads();
    }

    // ---- spill accumulators (+b2) to LDS; C layout: VGPR j = row j / j+8 ----
    {
        const int rowB = waveM * 16 + ((lane >> 4) << 3);
        #pragma unroll
        for (int t = 0; t < 8; ++t) {
            const int col = waveN * 128 + t * 16 + l16;
            const float bv = b2[p * E_DIM + col];
            #pragma unroll
            for (int j = 0; j < 8; ++j)
                Cs[(rowB + j) * LDC + col] = acc[t][j] + bv;
        }
    }
    __syncthreads();

    // ---- fused LayerNorm over E=256: one wave per 8-row group ----
    float g[8], be[8];
    #pragma unroll
    for (int j = 0; j < 8; ++j) {
        g[j]  = gamma[lane * 8 + j];
        be[j] = beta[lane * 8 + j];
    }
    for (int r = wid * 8; r < wid * 8 + 8; ++r) {
        float v[8];
        float s = 0.f, sq = 0.f;
        #pragma unroll
        for (int j = 0; j < 8; ++j) {
            v[j] = Cs[r * LDC + lane * 8 + j];
            s  += v[j];
            sq += v[j] * v[j];
        }
        #pragma unroll
        for (int off = 16; off > 0; off >>= 1) {
            s  += __shfl_xor(s,  off, 32);
            sq += __shfl_xor(sq, off, 32);
        }
        const float mean = s * (1.0f / E_DIM);
        const float var  = sq * (1.0f / E_DIM) - mean * mean;
        const float rs   = rsqrtf(var + LN_EPS);
        float o[8];
        #pragma unroll
        for (int j = 0; j < 8; ++j)
            o[j] = (v[j] - mean) * rs * g[j] + be[j];
        float4* dst = (float4*)(out + ((size_t)(b0 + r) * T_TOK + p) * E_DIM + lane * 8);
        dst[0] = make_float4(o[0], o[1], o[2], o[3]);
        dst[1] = make_float4(o[4], o[5], o[6], o[7]);
    }
}

// Embedding gather + LayerNorm for categorical tokens: one wave per (b,q) row.
__global__ __launch_bounds__(256)
void cat_tok_kernel(const int* __restrict__ x_cat,
                    const float* __restrict__ emb,
                    const float* __restrict__ gamma,
                    const float* __restrict__ beta,
                    float* __restrict__ out) {
    const int tid  = threadIdx.x;
    const int lane = tid & 31;
    const int wid  = tid >> 5;
    const int row  = blockIdx.x * 8 + wid;      // over B*Q
    const int b = row / Q_CAT;
    const int q = row - b * Q_CAT;

    const int idx = x_cat[(size_t)b * Q_CAT + q];
    const float* src = emb + ((size_t)q * V_VOC + idx) * E_DIM + lane * 8;

    float v[8];
    float s = 0.f, sq = 0.f;
    const float4 v0 = ((const float4*)src)[0];
    const float4 v1 = ((const float4*)src)[1];
    v[0] = v0.x; v[1] = v0.y; v[2] = v0.z; v[3] = v0.w;
    v[4] = v1.x; v[5] = v1.y; v[6] = v1.z; v[7] = v1.w;
    #pragma unroll
    for (int j = 0; j < 8; ++j) { s += v[j]; sq += v[j] * v[j]; }
    #pragma unroll
    for (int off = 16; off > 0; off >>= 1) {
        s  += __shfl_xor(s,  off, 32);
        sq += __shfl_xor(sq, off, 32);
    }
    const float mean = s * (1.0f / E_DIM);
    const float var  = sq * (1.0f / E_DIM) - mean * mean;
    const float rs   = rsqrtf(var + LN_EPS);

    float o[8];
    #pragma unroll
    for (int j = 0; j < 8; ++j)
        o[j] = (v[j] - mean) * rs * gamma[lane * 8 + j] + beta[lane * 8 + j];

    float4* dst = (float4*)(out + ((size_t)b * T_TOK + P_NUM + q) * E_DIM + lane * 8);
    dst[0] = make_float4(o[0], o[1], o[2], o[3]);
    dst[1] = make_float4(o[4], o[5], o[6], o[7]);
}

extern "C" void kernel_launch(void* const* d_in, const int* in_sizes, int n_in,
                              void* d_out, int out_size, void* d_ws, size_t ws_size,
                              hipStream_t stream) {
    (void)in_sizes; (void)n_in; (void)out_size; (void)d_ws; (void)ws_size;
    const float* x_num = (const float*)d_in[0];
    const int*   x_cat = (const int*)d_in[1];
    const float* W1    = (const float*)d_in[2];
    const float* b1    = (const float*)d_in[3];
    const float* W2    = (const float*)d_in[4];
    const float* b2    = (const float*)d_in[5];
    const float* emb   = (const float*)d_in[6];
    const float* gamma = (const float*)d_in[7];
    const float* beta  = (const float*)d_in[8];
    float* out = (float*)d_out;

    dim3 g1(P_NUM, B_SZ / MT);   // 24 x 128 blocks
    num_tok_kernel<<<g1, 256, 0, stream>>>(x_num, W1, b1, W2, b2, gamma, beta, out);
    cat_tok_kernel<<<(B_SZ * Q_CAT) / 8, 256, 0, stream>>>(x_cat, emb, gamma, beta, out);
}
